// M304B_14508399526721
// MI455X (gfx1250) — matrix operand order repaired and checked by reading, not compiled, not run on hardware
//
#include <hip/hip_runtime.h>
#include <hip/hip_bf16.h>
#include <math.h>

// ---------------- problem constants (from reference) ----------------
#define NNODES 50000
#define NEDGES 850000          // 800000 neighbor edges + 50000 self loops
#define NGRAPH 100
#define DIM    128             // node feature dim D
#define DEDGE  64              // edge feature dim DE
#define DHID   256             // node MLP hidden DH
#define DPE    192             // D + DE
#define LN_EPS 1e-5f
#define GN_EPS 1e-5f
#define ROWT   3125            // NNODES / 16 (exact)

typedef __attribute__((ext_vector_type(16))) _Float16 v16h;
typedef __attribute__((ext_vector_type(8)))  _Float16 v8h;
typedef __attribute__((ext_vector_type(8)))  float    v8f;

__device__ __forceinline__ float silu_f(float x) { return x / (1.0f + __expf(-x)); }

__device__ __forceinline__ v8f wmma_fma(v16h a, v16h b, v8f c) {
  // D = A(16x32 f16) * B(32x16 f16) + C(16x16 f32)
  return __builtin_amdgcn_wmma_f32_16x16x32_f16(false, a, false, b, (short)0, c,
                                                false, false);
}

// A fragment: 16x32 f16 tile from row-major [rows x lda], contiguous k order.
// lanes 0-15: row=lane, elems 0..15 = K 0..15
// lanes16-31: row=lane-16, elems 0..15 = K 16..31
// 16 contiguous halves = 32 bytes per lane -> one v16h load.
__device__ __forceinline__ v16h ldA(const _Float16* A, int lda, int row0, int k0, int lane) {
  const _Float16* p = A + (size_t)(row0 + (lane & 15)) * lda + (k0 + ((lane & 16) ? 16 : 0));
  return *(const v16h*)p;
}

// B fragment from TRANSPOSED weights WT[Nout][K] (row-major, ldk = K).
// lanes 0-15: col=lane, elems = K k0..k0+15 ; lanes16-31: col=lane-16, K +16..+31
// 16 contiguous halves = 32 bytes per lane -> one v16h load (2x b128).
__device__ __forceinline__ v16h ldBT(const _Float16* BT, int ldk, int k0, int col0, int lane) {
  const _Float16* p = BT + (size_t)(col0 + (lane & 15)) * ldk + (k0 + ((lane & 16) ? 16 : 0));
  return *(const v16h*)p;
}

// ---------------- small utility kernels ----------------
__global__ void k_zero_f32(float* p, int n) {
  int i = blockIdx.x * blockDim.x + threadIdx.x;
  if (i < n) p[i] = 0.0f;
}

// convert f32 W[K][N] (row-major) -> f16 WT[N][K] (row-major, transposed)
__global__ void k_f32_to_f16_T(const float* a, _Float16* bt, int K, int N) {
  int i = blockIdx.x * blockDim.x + threadIdx.x;
  if (i >= K * N) return;
  int k = i / N, n = i - k * N;
  bt[(size_t)n * K + k] = (_Float16)a[i];
}

__global__ void k_count_nodes(const int* batch, float* gcnt, int n) {
  int i = blockIdx.x * blockDim.x + threadIdx.x;
  if (i < n) atomicAdd(&gcnt[batch[i]], 1.0f);
}

// normalized edge length scalar per edge
__global__ void k_edge_scalar(const float* pos, const int* src, const int* dst,
                              float* eo, int E) {
  int i = blockIdx.x * blockDim.x + threadIdx.x;
  if (i >= E) return;
  int s = src[i], t = dst[i];
  float dx = pos[s * 3 + 0] - pos[t * 3 + 0];
  float dy = pos[s * 3 + 1] - pos[t * 3 + 1];
  float dz = pos[s * 3 + 2] - pos[t * 3 + 2];
  float l2 = dx * dx + dy * dy + dz * dz;
  float el = sqrtf(fmaxf(l2, 1e-12f));
  eo[i] = (el - 2.7554f) * (1.0f / 1.1664f);
}

__global__ void k_embed(const float* emb, const int* atom, float* h, _Float16* h16) {
  int i = blockIdx.x * blockDim.x + threadIdx.x;
  if (i >= NNODES * DIM) return;
  int n = i >> 7, c = i & 127;
  float v = emb[atom[n] * DIM + c];
  h[i] = v; h16[i] = (_Float16)v;
}

// ---------------- fused edge kernel ----------------
// Per block: 64 edges. Phase1: edge MLP stage-0 (1->64, LN, SiLU) into LDS (f16).
// Phase2: WMMA GEMM t(64x64) @ W1(64x64) + b1 into LDS (f32).
// Phase3: scatter-add message concat([h[src], ea]) into h0[dst] (192 wide).
#define EB 64
__global__ void k_edge_fused(const float* __restrict__ ev,
                             const int* __restrict__ src, const int* __restrict__ dst,
                             const float* __restrict__ w0b, const float* __restrict__ w0w,
                             const float* __restrict__ lnb, const float* __restrict__ lng,
                             const _Float16* __restrict__ w1T, const float* __restrict__ w1b,
                             const float* __restrict__ h, float* __restrict__ h0, int E) {
  __shared__ _Float16 tA[EB * DEDGE];   // 8 KB: LN+SiLU edge activations (f16)
  __shared__ float    ea[EB * DEDGE];   // 16 KB: edge MLP output (f32)
  const int tid = threadIdx.x;          // blockDim = 128 (4 waves)
  const int e0 = blockIdx.x * EB;

  if (tid < EB) {
    int ei = e0 + tid;
    if (ei < E) {
      float e = ev[ei];
      float s = 0.0f, s2 = 0.0f;
#pragma unroll 8
      for (int j = 0; j < DEDGE; ++j) {
        float v = fmaf(e, w0w[j], w0b[j]);
        s += v; s2 += v * v;
      }
      float m = s * (1.0f / DEDGE);
      float var = fmaxf(s2 * (1.0f / DEDGE) - m * m, 0.0f);
      float rs = rsqrtf(var + LN_EPS);
#pragma unroll 8
      for (int j = 0; j < DEDGE; ++j) {
        float v = fmaf(e, w0w[j], w0b[j]);
        v = (v - m) * rs * lng[j] + lnb[j];
        tA[tid * DEDGE + j] = (_Float16)silu_f(v);
      }
    } else {
#pragma unroll 8
      for (int j = 0; j < DEDGE; ++j) tA[tid * DEDGE + j] = (_Float16)0.0f;
    }
  }
  __syncthreads();

  // phase 2: WMMA. wave w owns edge rows [16w,16w+16), all 4 col tiles
  const int wave = tid >> 5, lane = tid & 31;
  const int row0 = wave * 16;
#pragma unroll
  for (int ct = 0; ct < 4; ++ct) {
    v8f acc = {};
#pragma unroll
    for (int kk = 0; kk < DEDGE; kk += 32) {
      v16h a = ldA(tA, DEDGE, row0, kk, lane);
      v16h b = ldBT(w1T, DEDGE, kk, ct * 16, lane);
      acc = wmma_fma(a, b, acc);
    }
    int nn = lane & 15, rb = (lane & 16) ? 8 : 0;
    float bias = w1b[ct * 16 + nn];
#pragma unroll
    for (int i = 0; i < 8; ++i)
      ea[(row0 + rb + i) * DEDGE + ct * 16 + nn] = acc[i] + bias;
  }
  __syncthreads();

  // phase 3: scatter messages (192 per edge = 128 from h[src] + 64 edge attrs)
  for (int k = tid; k < EB * DPE; k += 128) {
    int le = k / DPE, j = k - le * DPE;
    int ei = e0 + le;
    if (ei < E) {
      int dn = dst[ei];
      float val = (j < DIM) ? h[(size_t)src[ei] * DIM + j] : ea[le * DEDGE + (j - DIM)];
      atomicAdd(&h0[(size_t)dn * DPE + j], val);
    }
  }
}

// ---------------- node GEMM kernels ----------------
// zp = h0 + h @ Wp + bp  -> f16 [N x 192]. grid(ROWT, 3), block 128.
__global__ void k_proj_gemm(const _Float16* __restrict__ h16, const _Float16* __restrict__ wpT,
                            const float* __restrict__ bp, const float* __restrict__ h0,
                            _Float16* __restrict__ zp16) {
  const int wave = threadIdx.x >> 5, lane = threadIdx.x & 31;
  const int row0 = blockIdx.x * 16;
  const int c0 = (blockIdx.y * 4 + wave) * 16;
  v8f acc = {};
#pragma unroll
  for (int k = 0; k < DIM; k += 32) {
    if (k + 32 < DIM) __builtin_prefetch(&wpT[(size_t)c0 * DIM + k + 32], 0, 1);
    v16h a = ldA(h16, DIM, row0, k, lane);
    v16h b = ldBT(wpT, DIM, k, c0, lane);
    acc = wmma_fma(a, b, acc);
  }
  int nn = lane & 15, rb = (lane & 16) ? 8 : 0;
  float bias = bp[c0 + nn];
#pragma unroll
  for (int i = 0; i < 8; ++i) {
    size_t r = row0 + rb + i;
    zp16[r * DPE + c0 + nn] = (_Float16)(acc[i] + bias + h0[r * DPE + c0 + nn]);
  }
}

// z1 = silu(LN(zp @ W0 + b0)) -> f16 [N x 256]. grid ROWT, block 256 (8 waves).
__global__ void k_mlp0_fused(const _Float16* __restrict__ zp16, const _Float16* __restrict__ w0T,
                             const float* __restrict__ b0, const float* __restrict__ lng,
                             const float* __restrict__ lnb, _Float16* __restrict__ z1_16) {
  __shared__ float zb[16][DHID + 1];
  __shared__ float mrow[16], rsrow[16];
  const int wave = threadIdx.x >> 5, lane = threadIdx.x & 31;
  const int row0 = blockIdx.x * 16;
#pragma unroll
  for (int t = 0; t < 2; ++t) {
    int c0 = (wave * 2 + t) * 16;
    v8f acc = {};
#pragma unroll
    for (int k = 0; k < DPE; k += 32) {
      if (k + 32 < DPE) __builtin_prefetch(&w0T[(size_t)c0 * DPE + k + 32], 0, 1);
      v16h a = ldA(zp16, DPE, row0, k, lane);
      v16h b = ldBT(w0T, DPE, k, c0, lane);
      acc = wmma_fma(a, b, acc);
    }
    int nn = lane & 15, rb = (lane & 16) ? 8 : 0;
    float bias = b0[c0 + nn];
#pragma unroll
    for (int i = 0; i < 8; ++i) zb[rb + i][c0 + nn] = acc[i] + bias;
  }
  __syncthreads();
  if (threadIdx.x < 16) {
    int r = threadIdx.x;
    float s = 0.0f, s2 = 0.0f;
    for (int c = 0; c < DHID; ++c) { float v = zb[r][c]; s += v; s2 += v * v; }
    float m = s * (1.0f / DHID);
    float var = fmaxf(s2 * (1.0f / DHID) - m * m, 0.0f);
    mrow[r] = m; rsrow[r] = rsqrtf(var + LN_EPS);
  }
  __syncthreads();
  for (int k = threadIdx.x; k < 16 * DHID; k += 256) {
    int r = k >> 8, c = k & 255;
    float v = (zb[r][c] - mrow[r]) * rsrow[r] * lng[c] + lnb[c];
    z1_16[(size_t)(row0 + r) * DHID + c] = (_Float16)silu_f(v);
  }
}

// z2 = z1 @ W1 + b1; then LN(node_ln1)+SiLU; store h; accumulate per-graph sums.
__global__ void k_mlp1_gn_pre(const _Float16* __restrict__ z1_16, const _Float16* __restrict__ w1T,
                              const float* __restrict__ b1, const float* __restrict__ lng,
                              const float* __restrict__ lnb, const int* __restrict__ batch,
                              float* __restrict__ hout, float* __restrict__ gsum) {
  __shared__ float zb[16][DIM + 1];
  __shared__ float mrow[16], rsrow[16];
  const int wave = threadIdx.x >> 5, lane = threadIdx.x & 31;
  const int row0 = blockIdx.x * 16;
  const int c0 = wave * 16;
  v8f acc = {};
#pragma unroll
  for (int k = 0; k < DHID; k += 32) {
    if (k + 32 < DHID) __builtin_prefetch(&w1T[(size_t)c0 * DHID + k + 32], 0, 1);
    v16h a = ldA(z1_16, DHID, row0, k, lane);
    v16h b = ldBT(w1T, DHID, k, c0, lane);
    acc = wmma_fma(a, b, acc);
  }
  int nn = lane & 15, rb = (lane & 16) ? 8 : 0;
  float bias = b1[c0 + nn];
#pragma unroll
  for (int i = 0; i < 8; ++i) zb[rb + i][c0 + nn] = acc[i] + bias;
  __syncthreads();
  if (threadIdx.x < 16) {
    int r = threadIdx.x;
    float s = 0.0f, s2 = 0.0f;
    for (int c = 0; c < DIM; ++c) { float v = zb[r][c]; s += v; s2 += v * v; }
    float m = s * (1.0f / DIM);
    float var = fmaxf(s2 * (1.0f / DIM) - m * m, 0.0f);
    mrow[r] = m; rsrow[r] = rsqrtf(var + LN_EPS);
  }
  __syncthreads();
  for (int k = threadIdx.x; k < 16 * DIM; k += 256) {
    int r = k >> 7, c = k & 127;
    float v = (zb[r][c] - mrow[r]) * rsrow[r] * lng[c] + lnb[c];
    v = silu_f(v);
    int node = row0 + r;
    hout[(size_t)node * DIM + c] = v;
    atomicAdd(&gsum[(size_t)batch[node] * DIM + c], v);
  }
}

// last layer: z2 = z1 @ W1 + b1; accumulate per-graph sum of row means.
__global__ void k_mlp1_last(const _Float16* __restrict__ z1_16, const _Float16* __restrict__ w1T,
                            const float* __restrict__ b1, const int* __restrict__ batch,
                            float* __restrict__ outacc) {
  __shared__ float zb[16][DIM + 1];
  const int wave = threadIdx.x >> 5, lane = threadIdx.x & 31;
  const int row0 = blockIdx.x * 16;
  const int c0 = wave * 16;
  v8f acc = {};
#pragma unroll
  for (int k = 0; k < DHID; k += 32) {
    v16h a = ldA(z1_16, DHID, row0, k, lane);
    v16h b = ldBT(w1T, DHID, k, c0, lane);
    acc = wmma_fma(a, b, acc);
  }
  int nn = lane & 15, rb = (lane & 16) ? 8 : 0;
  float bias = b1[c0 + nn];
#pragma unroll
  for (int i = 0; i < 8; ++i) zb[rb + i][c0 + nn] = acc[i] + bias;
  __syncthreads();
  if (threadIdx.x < 16) {
    int r = threadIdx.x;
    float s = 0.0f;
    for (int c = 0; c < DIM; ++c) s += zb[r][c];
    atomicAdd(&outacc[batch[row0 + r]], s * (1.0f / DIM));
  }
}

// ---------------- GraphNorm helpers ----------------
__global__ void k_seg_div(float* s, const float* cnt, int d, int n) {
  int i = blockIdx.x * blockDim.x + threadIdx.x;
  if (i < n) s[i] /= fmaxf(cnt[i / d], 1.0f);
}

__global__ void k_gn_center(float* h, const float* m, const float* gms,
                            const int* batch, float* gsum2) {
  int i = blockIdx.x * blockDim.x + threadIdx.x;
  if (i >= NNODES * DIM) return;
  int n = i >> 7, c = i & 127;
  float v = h[i] - gms[c] * m[(size_t)batch[n] * DIM + c];
  h[i] = v;
  atomicAdd(&gsum2[(size_t)batch[n] * DIM + c], v * v);
}

__global__ void k_gn_final(float* h, _Float16* h16, const float* var,
                           const float* g, const float* b, const int* batch) {
  int i = blockIdx.x * blockDim.x + threadIdx.x;
  if (i >= NNODES * DIM) return;
  int n = i >> 7, c = i & 127;
  float z = g[c] * h[i] * rsqrtf(var[(size_t)batch[n] * DIM + c] + GN_EPS) + b[c];
  z = silu_f(z);
  h[i] = z; h16[i] = (_Float16)z;
}

__global__ void k_finalize(const float* outacc, const float* cnt, float* out) {
  int g = blockIdx.x * blockDim.x + threadIdx.x;
  if (g < NGRAPH) out[g] = outacc[g] / fmaxf(cnt[g], 1.0f);
}

// ---------------- host side ----------------
struct LayerP {
  const float *e0b, *e0w, *e1b, *e1w, *ln0b, *ln0g;
  const float *gnb, *gng, *gnms;
  const float *n0b, *n0w, *n1b, *n1w, *nln0b, *nln0g, *nln1b, *nln1g;
  const float *pb, *pw;
};

extern "C" void kernel_launch(void* const* d_in, const int* in_sizes, int n_in,
                              void* d_out, int out_size, void* d_ws, size_t ws_size,
                              hipStream_t stream) {
  (void)in_sizes; (void)n_in; (void)out_size; (void)ws_size;
  // ---- unpack inputs: pytree order (sorted dict keys, lists in order) ----
  int ix = 0;
  const float* pos = (const float*)d_in[ix++];            // [N,3]
  const float* emb = (const float*)d_in[ix++];            // params.emb [200,128]
  LayerP L[3];
  for (int l = 0; l < 3; ++l) {
    bool last = (l == 2);
    L[l].e0b  = (const float*)d_in[ix++];  L[l].e0w  = (const float*)d_in[ix++];
    L[l].e1b  = (const float*)d_in[ix++];  L[l].e1w  = (const float*)d_in[ix++];
    L[l].ln0b = (const float*)d_in[ix++];  L[l].ln0g = (const float*)d_in[ix++];
    if (!last) {
      L[l].gnb  = (const float*)d_in[ix++];
      L[l].gng  = (const float*)d_in[ix++];
      L[l].gnms = (const float*)d_in[ix++];
    } else { L[l].gnb = L[l].gng = L[l].gnms = nullptr; }
    L[l].n0b   = (const float*)d_in[ix++]; L[l].n0w   = (const float*)d_in[ix++];
    L[l].n1b   = (const float*)d_in[ix++]; L[l].n1w   = (const float*)d_in[ix++];
    L[l].nln0b = (const float*)d_in[ix++]; L[l].nln0g = (const float*)d_in[ix++];
    if (!last) {
      L[l].nln1b = (const float*)d_in[ix++]; L[l].nln1g = (const float*)d_in[ix++];
    } else { L[l].nln1b = L[l].nln1g = nullptr; }
    L[l].pb = (const float*)d_in[ix++];    L[l].pw = (const float*)d_in[ix++];
  }
  const int* atom  = (const int*)d_in[ix++];
  const int* batch = (const int*)d_in[ix++];
  const int* eidx  = (const int*)d_in[ix++];              // [2, E]
  const int* src = eidx;
  const int* dst = eidx + NEDGES;
  float* out = (float*)d_out;

  // ---- carve workspace ----
  char* w = (char*)d_ws;
  auto carve = [&](size_t bytes) -> void* {
    void* p = (void*)w; w += (bytes + 255) & ~(size_t)255; return p;
  };
  float*    e_ws  = (float*)   carve((size_t)NEDGES * 4);
  float*    h     = (float*)   carve((size_t)NNODES * DIM * 4);
  _Float16* h16   = (_Float16*)carve((size_t)NNODES * DIM * 2);
  float*    h0    = (float*)   carve((size_t)NNODES * DPE * 4);
  _Float16* zp16  = (_Float16*)carve((size_t)NNODES * DPE * 2);
  _Float16* z1_16 = (_Float16*)carve((size_t)NNODES * DHID * 2);
  float*    gsum  = (float*)   carve((size_t)NGRAPH * DIM * 4);
  float*    gsum2 = (float*)   carve((size_t)NGRAPH * DIM * 4);
  float*    gcnt  = (float*)   carve((size_t)NGRAPH * 4);
  float*    outac = (float*)   carve((size_t)NGRAPH * 4);
  // transposed f16 weights: WT[Nout][K]
  _Float16* w1eT  = (_Float16*)carve((size_t)DEDGE * DEDGE * 2);  // [64][64]
  _Float16* wpT   = (_Float16*)carve((size_t)DPE * DIM * 2);      // [192][128]
  _Float16* w0T   = (_Float16*)carve((size_t)DHID * DPE * 2);     // [256][192]
  _Float16* w1T   = (_Float16*)carve((size_t)DIM * DHID * 2);     // [128][256]

  auto blocks = [](int n, int b) { return (n + b - 1) / b; };

  // ---- prologue ----
  k_zero_f32<<<blocks(NGRAPH, 128), 128, 0, stream>>>(gcnt, NGRAPH);
  k_zero_f32<<<blocks(NGRAPH, 128), 128, 0, stream>>>(outac, NGRAPH);
  k_count_nodes<<<blocks(NNODES, 256), 256, 0, stream>>>(batch, gcnt, NNODES);
  k_edge_scalar<<<blocks(NEDGES, 256), 256, 0, stream>>>(pos, src, dst, e_ws, NEDGES);
  k_embed<<<blocks(NNODES * DIM, 256), 256, 0, stream>>>(emb, atom, h, h16);

  // ---- layers ----
  for (int l = 0; l < 3; ++l) {
    bool last = (l == 2);
    k_f32_to_f16_T<<<blocks(DEDGE * DEDGE, 256), 256, 0, stream>>>(L[l].e1w, w1eT, DEDGE, DEDGE);
    k_f32_to_f16_T<<<blocks(DIM * DPE, 256), 256, 0, stream>>>(L[l].pw, wpT, DIM, DPE);
    k_f32_to_f16_T<<<blocks(DPE * DHID, 256), 256, 0, stream>>>(L[l].n0w, w0T, DPE, DHID);
    k_f32_to_f16_T<<<blocks(DHID * DIM, 256), 256, 0, stream>>>(L[l].n1w, w1T, DHID, DIM);
    k_zero_f32<<<blocks(NNODES * DPE, 256), 256, 0, stream>>>(h0, NNODES * DPE);

    k_edge_fused<<<blocks(NEDGES, EB), 128, 0, stream>>>(
        e_ws, src, dst, L[l].e0b, L[l].e0w, L[l].ln0b, L[l].ln0g,
        w1eT, L[l].e1b, h, h0, NEDGES);

    k_proj_gemm<<<dim3(ROWT, 3), 128, 0, stream>>>(h16, wpT, L[l].pb, h0, zp16);
    k_mlp0_fused<<<ROWT, 256, 0, stream>>>(zp16, w0T, L[l].n0b, L[l].nln0g, L[l].nln0b, z1_16);

    if (!last) {
      k_zero_f32<<<blocks(NGRAPH * DIM, 256), 256, 0, stream>>>(gsum, NGRAPH * DIM);
      k_zero_f32<<<blocks(NGRAPH * DIM, 256), 256, 0, stream>>>(gsum2, NGRAPH * DIM);
      k_mlp1_gn_pre<<<ROWT, 256, 0, stream>>>(z1_16, w1T, L[l].n1b, L[l].nln1g,
                                              L[l].nln1b, batch, h, gsum);
      k_seg_div<<<blocks(NGRAPH * DIM, 256), 256, 0, stream>>>(gsum, gcnt, DIM, NGRAPH * DIM);
      k_gn_center<<<blocks(NNODES * DIM, 256), 256, 0, stream>>>(h, gsum, L[l].gnms, batch, gsum2);
      k_seg_div<<<blocks(NGRAPH * DIM, 256), 256, 0, stream>>>(gsum2, gcnt, DIM, NGRAPH * DIM);
      k_gn_final<<<blocks(NNODES * DIM, 256), 256, 0, stream>>>(h, h16, gsum2, L[l].gng,
                                                                L[l].gnb, batch);
    } else {
      k_mlp1_last<<<ROWT, 256, 0, stream>>>(z1_16, w1T, L[l].n1b, batch, outac);
    }
  }

  k_finalize<<<1, 128, 0, stream>>>(outac, gcnt, out);
}
